// stp_attention_pooling_36378372997402
// MI455X (gfx1250) — matrix operand order repaired and checked by reading, not compiled, not run on hardware
//
#include <hip/hip_runtime.h>
#include <hip/hip_bf16.h>

typedef __attribute__((ext_vector_type(16))) _Float16 v16h;
typedef __attribute__((ext_vector_type(8)))  _Float16 v8h;
typedef __attribute__((ext_vector_type(8)))  float    v8f;

#define NN 2048
#define HH 128
#define LL 20

__device__ __forceinline__ float relu_(float x) { return fmaxf(x, 0.0f); }

// 2->16->1 MLP with relu on both layers. w1 is (2,16) row-major, w2 is (16,1).
__device__ __forceinline__ float mlp2(float x, float y,
                                      const float* __restrict__ w1,
                                      const float* __restrict__ b1,
                                      const float* __restrict__ w2,
                                      const float* __restrict__ b2) {
    float acc = b2[0];
    #pragma unroll
    for (int k = 0; k < 16; ++k) {
        float h = relu_(fmaf(x, w1[k], fmaf(y, w1[16 + k], b1[k])));
        acc = fmaf(h, w2[k], acc);
    }
    return relu_(acc);
}

// ---------------------------------------------------------------------------
// Kernel 1: per-node precompute: ends, bf, abs_a = MLP_abs(ends)
// ---------------------------------------------------------------------------
__global__ void precompute_kernel(const float* __restrict__ gt,
                                  const float* __restrict__ abs_w1,
                                  const float* __restrict__ abs_b1,
                                  const float* __restrict__ abs_w2,
                                  const float* __restrict__ abs_b2,
                                  float* __restrict__ ends,
                                  float* __restrict__ bfv,
                                  float* __restrict__ abs_a) {
    int i = blockIdx.x * blockDim.x + threadIdx.x;
    if (i >= NN) return;
    const float* base = gt + (size_t)i * LL * 2;
    float ex = base[(LL - 1) * 2 + 0];
    float ey = base[(LL - 1) * 2 + 1];
    float bx = base[(LL - 2) * 2 + 0];
    float by = base[(LL - 2) * 2 + 1];
    ends[2 * i + 0] = ex;
    ends[2 * i + 1] = ey;
    bfv[2 * i + 0] = bx;
    bfv[2 * i + 1] = by;
    abs_a[i] = mlp2(ex, ey, abs_w1, abs_b1, abs_w2, abs_b2);
}

// ---------------------------------------------------------------------------
// Kernel 2: hidden_state (2048x128 f32) -> TRANSPOSED split f16 (128x2048).
// Transposing makes every lane's WMMA B-fragment K-contiguous in memory.
// ---------------------------------------------------------------------------
__global__ void convert_hidden_kernel(const float* __restrict__ hs,
                                      _Float16* __restrict__ thi,
                                      _Float16* __restrict__ tlo) {
    int idx = blockIdx.x * blockDim.x + threadIdx.x;
    if (idx >= NN * HH) return;
    int j = idx / HH;   // node index (K dim of the GEMM)
    int d = idx % HH;   // hidden feature (N dim of the GEMM)
    float v = hs[idx];
    _Float16 hi = (_Float16)v;
    size_t t = (size_t)d * NN + j;
    thi[t] = hi;
    tlo[t] = (_Float16)(v - (float)hi);
}

// ---------------------------------------------------------------------------
// Kernel 3: one block per row i. 256 threads x 8 columns each.
// Computes scores via rel/rel/att MLPs, row softmax, writes w as split f16.
// ---------------------------------------------------------------------------
__global__ void score_softmax_kernel(const float* __restrict__ ends,
                                     const float* __restrict__ bfv,
                                     const float* __restrict__ abs_a,
                                     const float* __restrict__ rel_w1,
                                     const float* __restrict__ rel_b1,
                                     const float* __restrict__ rel_w2,
                                     const float* __restrict__ rel_b2,
                                     const float* __restrict__ att_w1,
                                     const float* __restrict__ att_b1,
                                     const float* __restrict__ att_w2,
                                     const float* __restrict__ att_b2,
                                     _Float16* __restrict__ w_hi,
                                     _Float16* __restrict__ w_lo) {
    __shared__ float red[256];
    const int i = blockIdx.x;
    const int tid = threadIdx.x;

    const float exi = ends[2 * i + 0], eyi = ends[2 * i + 1];
    const float bxi = bfv[2 * i + 0],  byi = bfv[2 * i + 1];
    const float ai  = abs_a[i];

    float s[8];
    #pragma unroll
    for (int k = 0; k < 8; ++k) {
        int j = tid + 256 * k;
        float repx = ends[2 * j + 0] - exi;
        float repy = ends[2 * j + 1] - eyi;
        float rvx  = repx - (bfv[2 * j + 0] - bxi);
        float rvy  = repy - (bfv[2 * j + 1] - byi);
        float rel_a = mlp2(repx, repy, rel_w1, rel_b1, rel_w2, rel_b2);
        float rv_a  = mlp2(rvx,  rvy,  rel_w1, rel_b1, rel_w2, rel_b2);
        float aj = abs_a[j];
        // att MLP: inputs (rel_a, rv_a, abs_a[j], abs_a[i]); att_w1 is (4,16)
        float acc = att_b2[0];
        #pragma unroll
        for (int q = 0; q < 16; ++q) {
            float h = fmaf(rel_a, att_w1[q],
                      fmaf(rv_a,  att_w1[16 + q],
                      fmaf(aj,    att_w1[32 + q],
                      fmaf(ai,    att_w1[48 + q], att_b1[q]))));
            acc = fmaf(relu_(h), att_w2[q], acc);
        }
        s[k] = relu_(acc);
    }

    // row max
    float m = s[0];
    #pragma unroll
    for (int k = 1; k < 8; ++k) m = fmaxf(m, s[k]);
    red[tid] = m;
    __syncthreads();
    for (int o = 128; o > 0; o >>= 1) {
        if (tid < o) red[tid] = fmaxf(red[tid], red[tid + o]);
        __syncthreads();
    }
    m = red[0];
    __syncthreads();

    // exp + row sum
    float lsum = 0.0f;
    #pragma unroll
    for (int k = 0; k < 8; ++k) {
        s[k] = __expf(s[k] - m);
        lsum += s[k];
    }
    red[tid] = lsum;
    __syncthreads();
    for (int o = 128; o > 0; o >>= 1) {
        if (tid < o) red[tid] += red[tid + o];
        __syncthreads();
    }
    float inv = 1.0f / red[0];

    #pragma unroll
    for (int k = 0; k < 8; ++k) {
        int j = tid + 256 * k;
        float wv = s[k] * inv;
        _Float16 hi = (_Float16)wv;
        size_t idx = (size_t)i * NN + j;
        w_hi[idx] = hi;
        w_lo[idx] = (_Float16)(wv - (float)hi);
    }
}

// ---------------------------------------------------------------------------
// Kernel 4: out = W(2048x2048) @ Hid(2048x128) via v_wmma_f32_16x16x32_f16.
// Split-f16: out = Whi*Thi + Whi*Tlo + Wlo*Thi (f32-level accuracy).
// One wave per 16x16 output tile; 4 waves per 128-thread block; 1024 tiles.
// A-fragment: per lane two K-contiguous 8-half chunks of row (rowBase+mr):
//   elements 0..7  <- K = kb + 16*half  .. +7
//   elements 8..15 <- K = kb + 16*half + 8 .. +7   (matches B's contiguous K order)
// B-fragment: per lane one K-contiguous 16-half chunk of Ht row (nBase+mr):
//   elements 0..15 <- K = kb + 16*half .. +15      (B 32x16, N=lane&15)
// EXEC is all-ones throughout (no divergence), as WMMA requires.
// ---------------------------------------------------------------------------
__global__ void wmma_gemm_kernel(const _Float16* __restrict__ Whi,
                                 const _Float16* __restrict__ Wlo,
                                 const _Float16* __restrict__ Thi,
                                 const _Float16* __restrict__ Tlo,
                                 float* __restrict__ out) {
    const int lane = threadIdx.x & 31;
    const int wave = threadIdx.x >> 5;
    const int tile = blockIdx.x * 4 + wave;      // 0..1023
    const int tm = tile >> 3;                    // 0..127 (row tile)
    const int tn = tile & 7;                     // 0..7   (col tile)
    const int rowBase = tm * 16;
    const int nBase = tn * 16;
    const int half = lane >> 4;
    const int mr = lane & 15;

    const _Float16* aHi = Whi + (size_t)(rowBase + mr) * NN + 16 * half;
    const _Float16* aLo = Wlo + (size_t)(rowBase + mr) * NN + 16 * half;
    const _Float16* bHi = Thi + (size_t)(nBase + mr) * NN + 16 * half;
    const _Float16* bLo = Tlo + (size_t)(nBase + mr) * NN + 16 * half;

    v8f c = {};
    for (int kb = 0; kb < NN; kb += 32) {
        if (kb + 32 < NN) {
            __builtin_prefetch(aHi + kb + 32, 0, 0);
            __builtin_prefetch(bHi + kb + 32, 0, 0);
        }
        v8h a0 = *(const v8h*)(aHi + kb);
        v8h a1 = *(const v8h*)(aHi + kb + 8);
        v8h l0 = *(const v8h*)(aLo + kb);
        v8h l1 = *(const v8h*)(aLo + kb + 8);
        v16h ahi = __builtin_shufflevector(a0, a1, 0, 1, 2, 3, 4, 5, 6, 7,
                                           8, 9, 10, 11, 12, 13, 14, 15);
        v16h alo = __builtin_shufflevector(l0, l1, 0, 1, 2, 3, 4, 5, 6, 7,
                                           8, 9, 10, 11, 12, 13, 14, 15);
        v16h bhi = *(const v16h*)(bHi + kb);
        v16h blo = *(const v16h*)(bLo + kb);

        c = __builtin_amdgcn_wmma_f32_16x16x32_f16(false, ahi, false, bhi,
                                                   (short)0, c, false, false);
        c = __builtin_amdgcn_wmma_f32_16x16x32_f16(false, ahi, false, blo,
                                                   (short)0, c, false, false);
        c = __builtin_amdgcn_wmma_f32_16x16x32_f16(false, alo, false, bhi,
                                                   (short)0, c, false, false);
    }

    // C/D layout: lanes 0-15: M=r, N=lane; lanes 16-31: M=r+8, N=lane-16
    #pragma unroll
    for (int r = 0; r < 8; ++r) {
        out[(size_t)(rowBase + r + 8 * half) * HH + (nBase + mr)] = c[r];
    }
}

// ---------------------------------------------------------------------------
extern "C" void kernel_launch(void* const* d_in, const int* in_sizes, int n_in,
                              void* d_out, int out_size, void* d_ws, size_t ws_size,
                              hipStream_t stream) {
    const float* hidden  = (const float*)d_in[0];
    const float* gt      = (const float*)d_in[1];
    const float* rel_w1  = (const float*)d_in[2];
    const float* rel_b1  = (const float*)d_in[3];
    const float* rel_w2  = (const float*)d_in[4];
    const float* rel_b2  = (const float*)d_in[5];
    const float* abs_w1  = (const float*)d_in[6];
    const float* abs_b1  = (const float*)d_in[7];
    const float* abs_w2  = (const float*)d_in[8];
    const float* abs_b2  = (const float*)d_in[9];
    const float* att_w1  = (const float*)d_in[10];
    const float* att_b1  = (const float*)d_in[11];
    const float* att_w2  = (const float*)d_in[12];
    const float* att_b2  = (const float*)d_in[13];
    float* out = (float*)d_out;

    // Workspace carve-up (256-byte aligned slices)
    char* ws = (char*)d_ws;
    size_t off = 0;
    auto carve = [&](size_t bytes) {
        char* p = ws + off;
        off += (bytes + 255) & ~(size_t)255;
        return p;
    };
    float*    ends  = (float*)carve((size_t)NN * 2 * sizeof(float));
    float*    bfv   = (float*)carve((size_t)NN * 2 * sizeof(float));
    float*    abs_a = (float*)carve((size_t)NN * sizeof(float));
    _Float16* thi   = (_Float16*)carve((size_t)NN * HH * sizeof(_Float16));
    _Float16* tlo   = (_Float16*)carve((size_t)NN * HH * sizeof(_Float16));
    _Float16* whi   = (_Float16*)carve((size_t)NN * NN * sizeof(_Float16));
    _Float16* wlo   = (_Float16*)carve((size_t)NN * NN * sizeof(_Float16));
    (void)ws_size; (void)in_sizes; (void)n_in; (void)out_size;

    precompute_kernel<<<NN / 256, 256, 0, stream>>>(
        gt, abs_w1, abs_b1, abs_w2, abs_b2, ends, bfv, abs_a);

    convert_hidden_kernel<<<(NN * HH) / 256, 256, 0, stream>>>(hidden, thi, tlo);

    score_softmax_kernel<<<NN, 256, 0, stream>>>(
        ends, bfv, abs_a,
        rel_w1, rel_b1, rel_w2, rel_b2,
        att_w1, att_b1, att_w2, att_b2,
        whi, wlo);

    // 1024 tiles of 16x16 over the 2048x128 output; 4 waves/block
    wmma_gemm_kernel<<<256, 128, 0, stream>>>(whi, wlo, thi, tlo, out);
}
